// MAB_6691559047216
// MI455X (gfx1250) — matrix operand order repaired and checked by reading, not compiled, not run on hardware
//
#include <hip/hip_runtime.h>
#include <hip/hip_bf16.h>

typedef __attribute__((ext_vector_type(16))) _Float16 v16h;
typedef __attribute__((ext_vector_type(8)))  float    v8f;
typedef __attribute__((ext_vector_type(4)))  unsigned int v4u;
typedef __attribute__((ext_vector_type(4)))  int v4i;

typedef __attribute__((address_space(1))) v4i gv4i_t;   // global int4
typedef __attribute__((address_space(3))) v4i lv4i_t;   // LDS int4

#define BB   4
#define NQ   2048
#define NK   2048
#define DD   256
#define HH   4
#define DH   64
#define MTOT (BB * NQ)   // 8192 rows

union frag16 { v16h v; v4u u[2]; };

__device__ __forceinline__ v8f wmma16(v16h a, v16h b, v8f c) {
  return __builtin_amdgcn_wmma_f32_16x16x32_f16(false, a, false, b, (short)0, c,
                                                false, false);
}

// 16-byte global -> LDS copy. Uses the CDNA5 async path (ASYNCcnt-tracked,
// bypasses VGPRs) when the builtin is declared, else a plain copy.
__device__ __forceinline__ void cp_g2l_16B(const _Float16* __restrict__ gsrc,
                                           _Float16* ldst) {
#if __has_builtin(__builtin_amdgcn_global_load_async_to_lds_b128)
  __builtin_amdgcn_global_load_async_to_lds_b128((gv4i_t*)gsrc, (lv4i_t*)ldst,
                                                 0, 0);
#else
  *(v4u*)ldst = *(const v4u*)gsrc;
#endif
}

__device__ __forceinline__ void wait_async0() {
#if __has_builtin(__builtin_amdgcn_global_load_async_to_lds_b128)
#if __has_builtin(__builtin_amdgcn_s_wait_asynccnt)
  __builtin_amdgcn_s_wait_asynccnt(0);
#else
  asm volatile("s_wait_asynccnt 0x0" ::: "memory");
#endif
#endif
}

// A fragment (16x32 f16): lane L holds row M=m_off+(L&15); K elems:
//   elements 0..7 -> Kbase..Kbase+7, elements 8..15 -> Kbase+16..Kbase+23,
//   Kbase=(L>>4)*8.  LDS row-major, `stride` halves/row (multiple of 8).
__device__ __forceinline__ v16h load_afrag(const _Float16* S, int stride,
                                           int m_off, int lane) {
  const _Float16* p = S + (m_off + (lane & 15)) * stride + ((lane >> 4) * 8);
  frag16 r;
  r.u[0] = *(const v4u*)p;
  r.u[1] = *(const v4u*)(p + 16);
  return r.v;
}

// B fragment (32x16 f16), contraction chunk c (K = 32c..32c+31): lane L holds
// column N=n_off+(L&15); elements 0..7 -> kb..kb+7, 8..15 -> kb+16..kb+23,
// kb=32c+(L>>4)*8.
// LDS layout [N][K], `stride` halves per N row.
__device__ __forceinline__ v16h load_bfrag(const _Float16* S, int stride,
                                           int n_off, int c, int lane) {
  const _Float16* p = S + (n_off + (lane & 15)) * stride + (c * 32 + (lane >> 4) * 8);
  frag16 r;
  r.u[0] = *(const v4u*)p;
  r.u[1] = *(const v4u*)(p + 16);
  return r.v;
}

// ---------------------------------------------------------------------------
// C = A[M,256] @ W[256,256](slice) + bias, 128x64 block tile, 8 waves, each a
// 32x32 tile (2x2 WMMA accumulators).  Optional f32 / f16 / per-head
// transposed f16 outputs (CT[b][h][d][n], used for V).
// ---------------------------------------------------------------------------
template <bool A_HALF, bool RELU>
__global__ __launch_bounds__(256) void gemm256_kernel(
    const void* __restrict__ Av, const float* __restrict__ W,
    const float* __restrict__ bias, float* __restrict__ Cf,
    _Float16* __restrict__ Ch, _Float16* __restrict__ CT) {
  __shared__ _Float16 As[128 * 40];
  __shared__ _Float16 Bs[64 * 40];
  const int tid   = threadIdx.x;
  const int lane  = tid & 31;
  const int wid   = tid >> 5;
  const int waveM = wid & 3;   // 4 waves along M (4*32 = 128)
  const int waveN = wid >> 2;  // 2 waves along N (2*32 = 64)
  const int mBase = blockIdx.x * 128;
  const int nBase = blockIdx.y * 64;

  v8f acc[2][2] = {};

  for (int k0 = 0; k0 < 256; k0 += 32) {
    // stage A chunk (128 x 32) as f16
    if (A_HALF) {
      const _Float16* A = (const _Float16*)Av;
      for (int i = 0; i < 2; ++i) {             // 512 16B chunks / 256 thr
        int chunk = i * 256 + tid;
        int r = chunk >> 2, k8 = (chunk & 3) * 8;
        cp_g2l_16B(A + (size_t)(mBase + r) * 256 + k0 + k8, &As[r * 40 + k8]);
      }
    } else {
      const float* A = (const float*)Av;
      for (int i = 0; i < 16; ++i) {
        int idx = i * 256 + tid;
        int r = idx >> 5, k = idx & 31;
        As[r * 40 + k] = (_Float16)A[(size_t)(mBase + r) * 256 + k0 + k];
      }
    }
    // stage B chunk (32 x 64) transposed -> Bs[n][k], f32 -> f16
    for (int i = 0; i < 8; ++i) {
      int idx = i * 256 + tid;
      int k = idx >> 6, n = idx & 63;
      Bs[n * 40 + k] = (_Float16)W[(size_t)(k0 + k) * 256 + nBase + n];
    }
    if (A_HALF) wait_async0();
    __syncthreads();

    v16h af0 = load_afrag(As, 40, waveM * 32, lane);
    v16h af1 = load_afrag(As, 40, waveM * 32 + 16, lane);
    for (int nt = 0; nt < 2; ++nt) {
      v16h bf = load_bfrag(Bs, 40, waveN * 32 + nt * 16, 0, lane);
      acc[0][nt] = wmma16(af0, bf, acc[0][nt]);
      acc[1][nt] = wmma16(af1, bf, acc[1][nt]);
    }
    __syncthreads();
  }

  // epilogue: C layout — VGPR r holds M=r (lanes 0-15) / M=r+8 (lanes 16-31),
  // N = lane & 15.
  for (int mt = 0; mt < 2; ++mt) {
    for (int nt = 0; nt < 2; ++nt) {
      int col = nBase + waveN * 32 + nt * 16 + (lane & 15);
      float bv = bias[col];
      for (int r = 0; r < 8; ++r) {
        int row = mBase + waveM * 32 + mt * 16 + r + ((lane >> 4) << 3);
        float v = acc[mt][nt][r] + bv;
        if (RELU) v = fmaxf(v, 0.0f);
        if (Cf) Cf[(size_t)row * 256 + col] = v;
        if (Ch) Ch[(size_t)row * 256 + col] = (_Float16)v;
        if (CT) {
          int b = row >> 11, nin = row & 2047;
          int h = col >> 6,  d   = col & 63;
          CT[(((size_t)(b * HH + h) * DH + d) << 11) + nin] = (_Float16)v;
        }
      }
    }
  }
}

// ---------------------------------------------------------------------------
// Flash attention: grid = (NQ/64, B*H), block = 128 (4 waves).
// Wave owns 16 q-rows; K/V stream through LDS via async copies; online
// softmax (row-max by shuffle tree, row-sum by WMMA against all-ones B).
// ---------------------------------------------------------------------------
__global__ __launch_bounds__(128) void attn_kernel(
    const _Float16* __restrict__ Qh, const _Float16* __restrict__ Kh,
    const _Float16* __restrict__ Vt, float* __restrict__ AO) {
  __shared__ _Float16 Ks[64 * 72];        // [kidx][d]
  __shared__ _Float16 Vs[64 * 72];        // [d][kidx]
  __shared__ _Float16 Ps[4][16 * 72];     // per-wave P scratch [m][kidx]

  const int tid  = threadIdx.x;
  const int lane = tid & 31;
  const int wid  = tid >> 5;
  const int b    = (int)blockIdx.y >> 2;
  const int h    = (int)blockIdx.y & 3;
  const int q0   = (int)blockIdx.x * 64 + wid * 16;
  const size_t qrow = (size_t)b * NQ + q0;

  // resident Q fragments (16 rows x 64 head-dims)
  v16h qf[2];
  {
    int m = lane & 15;
    for (int c = 0; c < 2; ++c) {
      const _Float16* p = Qh + (qrow + m) * 256 + h * DH + c * 32 + ((lane >> 4) * 8);
      frag16 r;
      r.u[0] = *(const v4u*)p;
      r.u[1] = *(const v4u*)(p + 16);
      qf[c] = r.v;
    }
  }
  // all-ones B fragment: rowsum(P) = P @ ones via WMMA
  v16h onesf;
  for (int i = 0; i < 16; ++i) onesf[i] = (_Float16)1.0f;

  v8f o[4] = {};
  float mrow[8], lrow[8];
  for (int r = 0; r < 8; ++r) { mrow[r] = -1e30f; lrow[r] = 0.0f; }
  const float scale = 0.125f;  // 1/sqrt(DH)

  for (int kb0 = 0; kb0 < NK; kb0 += 64) {
    // stage K tile: Ks[kidx][d]  (async global->LDS, 16B per lane-op)
    for (int i = 0; i < 4; ++i) {
      int idx = i * 128 + tid;               // 512 16-byte chunks
      int r = idx >> 3, d8 = (idx & 7) * 8;
      cp_g2l_16B(Kh + ((size_t)b * NK + kb0 + r) * 256 + h * DH + d8,
                 &Ks[r * 72 + d8]);
    }
    // stage V tile: Vs[d][kidx] from pre-transposed Vt
    for (int i = 0; i < 4; ++i) {
      int idx = i * 128 + tid;
      int d = idx >> 3, i8 = (idx & 7) * 8;
      cp_g2l_16B(Vt + (((size_t)(b * HH + h) * DH + d) << 11) + kb0 + i8,
                 &Vs[d * 72 + i8]);
    }
    wait_async0();
    __syncthreads();

    // S = Q * K^T  (contraction over d = 64 -> two 32-chunks), raw scores
    v8f s[4];
    for (int nt = 0; nt < 4; ++nt) {
      v8f a = {};
      for (int c = 0; c < 2; ++c)
        a = wmma16(qf[c], load_bfrag(Ks, 72, nt * 16, c, lane), a);
      s[nt] = a;
    }

    // row max (unscaled) over lanes 0-15 / 16-31 of the C layout, then
    // online rescale factor alpha = exp(scale*(m_old - m_new)).
    float alpha[8];
    for (int r = 0; r < 8; ++r) {
      float v = fmaxf(fmaxf(s[0][r], s[1][r]), fmaxf(s[2][r], s[3][r]));
      for (int off = 1; off < 16; off <<= 1) v = fmaxf(v, __shfl_xor(v, off, 32));
      float mn = fmaxf(mrow[r], v);
      alpha[r] = __expf(scale * (mrow[r] - mn));
      mrow[r] = mn;
    }
    for (int nt = 0; nt < 4; ++nt)
      for (int r = 0; r < 8; ++r) o[nt][r] *= alpha[r];

    // P = exp(scale*(s - m)) -> f16, C layout -> A layout via per-wave LDS
    // round trip (same-wave DS ordering, no block barrier needed).
    _Float16* myP = &Ps[wid][0];
    int radd = (lane >> 4) << 3;
    for (int nt = 0; nt < 4; ++nt) {
      int coln = nt * 16 + (lane & 15);
      for (int r = 0; r < 8; ++r)
        myP[(r + radd) * 72 + coln] = (_Float16)__expf(scale * (s[nt][r] - mrow[r]));
    }
    v16h pf[2];
    for (int c = 0; c < 2; ++c) {
      const _Float16* p = myP + (lane & 15) * 72 + c * 32 + ((lane >> 4) * 8);
      frag16 rr;
      rr.u[0] = *(const v4u*)p;
      rr.u[1] = *(const v4u*)(p + 16);
      pf[c] = rr.v;
    }

    // row sums via WMMA against all-ones (replicated across columns)
    {
      v8f rs = {};
      rs = wmma16(pf[0], onesf, rs);
      rs = wmma16(pf[1], onesf, rs);
      for (int r = 0; r < 8; ++r) lrow[r] = lrow[r] * alpha[r] + rs[r];
    }

    // O += P * V  (contraction over kidx = 64 -> two 32-chunks)
    for (int nt = 0; nt < 4; ++nt)
      for (int c = 0; c < 2; ++c)
        o[nt] = wmma16(pf[c], load_bfrag(Vs, 72, nt * 16, c, lane), o[nt]);

    __syncthreads();
  }

  for (int r = 0; r < 8; ++r) lrow[r] = 1.0f / lrow[r];
  for (int nt = 0; nt < 4; ++nt) {
    int col = h * DH + nt * 16 + (lane & 15);
    for (int r = 0; r < 8; ++r) {
      int row = (int)qrow + r + ((lane >> 4) << 3);
      AO[(size_t)row * 256 + col] = o[nt][r] * lrow[r];
    }
  }
}

// ---------------------------------------------------------------------------
// out = LayerNorm(X + Y) * g + beta ; one wave per 256-wide row.
// ---------------------------------------------------------------------------
__global__ __launch_bounds__(256) void add_ln_kernel(
    const float* __restrict__ X, const float* __restrict__ Y,
    const float* __restrict__ g, const float* __restrict__ be,
    float* __restrict__ outF, _Float16* __restrict__ outH) {
  int row  = blockIdx.x * 8 + (threadIdx.x >> 5);
  int lane = threadIdx.x & 31;
  const float* px = X + (size_t)row * 256;
  const float* py = Y ? Y + (size_t)row * 256 : nullptr;

  float v[8];
  float s = 0.0f;
  for (int i = 0; i < 8; ++i) {
    int c = lane + i * 32;
    float t = px[c] + (py ? py[c] : 0.0f);
    v[i] = t;
    s += t;
  }
  for (int off = 1; off < 32; off <<= 1) s += __shfl_xor(s, off, 32);
  float mean = s * (1.0f / 256.0f);
  float var = 0.0f;
  for (int i = 0; i < 8; ++i) {
    float d = v[i] - mean;
    var += d * d;
  }
  for (int off = 1; off < 32; off <<= 1) var += __shfl_xor(var, off, 32);
  float rstd = rsqrtf(var * (1.0f / 256.0f) + 1e-5f);
  for (int i = 0; i < 8; ++i) {
    int c = lane + i * 32;
    float t = (v[i] - mean) * rstd * g[c] + be[c];
    if (outF) outF[(size_t)row * 256 + c] = t;
    if (outH) outH[(size_t)row * 256 + c] = (_Float16)t;
  }
}

extern "C" void kernel_launch(void* const* d_in, const int* in_sizes, int n_in,
                              void* d_out, int out_size, void* d_ws, size_t ws_size,
                              hipStream_t stream) {
  const float* Q   = (const float*)d_in[0];
  const float* K   = (const float*)d_in[1];
  const float* Wq  = (const float*)d_in[2];
  const float* bq  = (const float*)d_in[3];
  const float* Wk  = (const float*)d_in[4];
  const float* bk  = (const float*)d_in[5];
  const float* Wv  = (const float*)d_in[6];
  const float* bv  = (const float*)d_in[7];
  const float* W1  = (const float*)d_in[8];
  const float* b1  = (const float*)d_in[9];
  const float* W2  = (const float*)d_in[10];
  const float* b2  = (const float*)d_in[11];
  const float* g0  = (const float*)d_in[12];
  const float* be0 = (const float*)d_in[13];
  const float* g1  = (const float*)d_in[14];
  const float* be1 = (const float*)d_in[15];

  char* ws = (char*)d_ws;
  size_t off = 0;
  const size_t NF = (size_t)MTOT * DD;            // 8192*256
  float*    Qp = (float*)(ws + off);    off += NF * 4;  // Qp fp32 (residual)
  _Float16* Qh = (_Float16*)(ws + off); off += NF * 2;  // Qp fp16
  _Float16* Kh = (_Float16*)(ws + off); off += NF * 2;  // Kp fp16
  _Float16* Vt = (_Float16*)(ws + off); off += NF * 2;  // Vp fp16, [b][h][d][n]
  float*    AO = (float*)(ws + off);    off += NF * 4;  // attention out fp32
  float*    O  = (float*)(ws + off);    off += NF * 4;  // LN0 out fp32
  _Float16* Oh = (_Float16*)(ws + off); off += NF * 2;  // LN0 out fp16
  _Float16* H1 = (_Float16*)(ws + off); off += NF * 2;  // relu(O@W1+b1) fp16
  float*    F  = (float*)(ws + off);    off += NF * 4;  // H1@W2+b2 fp32

  dim3 gG(MTOT / 128, DD / 64), bG(256);
  // projections
  gemm256_kernel<false, false><<<gG, bG, 0, stream>>>(Q, Wq, bq, Qp, Qh, nullptr);
  gemm256_kernel<false, false><<<gG, bG, 0, stream>>>(K, Wk, bk, nullptr, Kh, nullptr);
  gemm256_kernel<false, false><<<gG, bG, 0, stream>>>(K, Wv, bv, nullptr, nullptr, Vt);
  // attention
  attn_kernel<<<dim3(NQ / 64, BB * HH), 128, 0, stream>>>(Qh, Kh, Vt, AO);
  // O = LN(Qp + AO)
  add_ln_kernel<<<MTOT / 8, 256, 0, stream>>>(Qp, AO, g0, be0, O, Oh);
  // FFN
  gemm256_kernel<true, true><<<gG, bG, 0, stream>>>(Oh, W1, b1, nullptr, H1, nullptr);
  gemm256_kernel<true, false><<<gG, bG, 0, stream>>>(H1, W2, b2, F, nullptr, nullptr);
  // out = LN(O + F)
  add_ln_kernel<<<MTOT / 8, 256, 0, stream>>>(O, F, g1, be1, (float*)d_out, nullptr);
}
